// TorchAttractorLanguageModel_54133767798922
// MI455X (gfx1250) — matrix operand order repaired and checked by reading, not compiled, not run on hardware
//
#include <hip/hip_runtime.h>

#define BD    2048
#define DD    512
#define VV    50257
#define VPAD  50272          // round V up to multiple of 16
#define NT16  (VPAD / 16)    // 3142 column tiles
#define NSTEP 12

#define DT_C       0.04f
#define CUBIC_C    0.008f
#define NOISE_C    0.001f
#define LAMBDA_C   0.1f
#define SIGSCALE_C 0.5f

typedef float  v2f  __attribute__((ext_vector_type(2)));
typedef float  v8f  __attribute__((ext_vector_type(8)));
typedef __attribute__((ext_vector_type(8)))  __bf16 v8bf;
typedef __attribute__((ext_vector_type(16))) __bf16 v16bf;

// -------- block-wide sum over 256 threads (8 wave32s) --------
__device__ __forceinline__ float block_sum(float v, float* red) {
  #pragma unroll
  for (int m = 16; m >= 1; m >>= 1) v += __shfl_xor(v, m, 32);
  const int wave = threadIdx.x >> 5;
  const int lane = threadIdx.x & 31;
  __syncthreads();
  if (lane == 0) red[wave] = v;
  __syncthreads();
  float s = 0.f;
  #pragma unroll
  for (int i = 0; i < 8; ++i) s += red[i];
  return s;
}

// -------- kernel 1: drive = beta*0.5*normalize(LN-normalized emb + gamma*ctx) --------
__global__ __launch_bounds__(256) void drive_kernel(
    const int* __restrict__ tok, const float* __restrict__ fast0,
    const float* __restrict__ emb, const float* __restrict__ beta_p,
    const float* __restrict__ gamma_p, float* __restrict__ drive)
{
  __shared__ float red[8];
  const int b = blockIdx.x;
  const int t = threadIdx.x;                 // 2 elements per thread
  const float* row = emb + (size_t)tok[b] * DD;
  float x0 = row[t], x1 = row[t + 256];
  float mu  = block_sum(x0 + x1, red) * (1.f / DD);
  float d0 = x0 - mu, d1 = x1 - mu;
  float var = block_sum(d0 * d0 + d1 * d1, red) * (1.f / DD);
  float inv = rsqrtf(var + 1e-5f);
  float e0 = d0 * inv, e1 = d1 * inv;
  float en = sqrtf(block_sum(e0 * e0 + e1 * e1, red));
  float bs = 1.f / fmaxf(en, 1e-12f);
  float f0 = fast0[(size_t)b * DD + t], f1 = fast0[(size_t)b * DD + t + 256];
  float fn = sqrtf(block_sum(f0 * f0 + f1 * f1, red));
  float cs = (*gamma_p) / (fn + 1e-6f);
  float s0 = e0 * bs + f0 * cs, s1 = e1 * bs + f1 * cs;
  float sn = sqrtf(block_sum(s0 * s0 + s1 * s1, red));
  float k  = (*beta_p) * SIGSCALE_C / (sn + 1e-6f);
  drive[(size_t)b * DD + t]       = k * s0;
  drive[(size_t)b * DD + t + 256] = k * s1;
}

// -------- kernel 2: one attractor step; full f32 via V_WMMA_F32_16X16X4_F32 --------
// One WG = 16 rows x 512 cols; 8 waves x 4 N-tiles. A tile staged in LDS.
__global__ __launch_bounds__(256) void step_kernel(
    const float* __restrict__ fin, float* __restrict__ fout,
    const float* __restrict__ diff, const float* __restrict__ drive,
    const float* __restrict__ nz)
{
  __shared__ float At[16 * DD];              // 32 KB fast tile
  __shared__ float rmean[16];
  __shared__ float rnorm[16];
  const int t = threadIdx.x;
  const int row0 = blockIdx.x * 16;

  const float4* src = (const float4*)(fin + (size_t)row0 * DD);
  float4* dst = (float4*)At;
  #pragma unroll
  for (int i = 0; i < 8; ++i) dst[t + i * 256] = src[t + i * 256];
  if (t < 16) { rmean[t] = 0.f; rnorm[t] = 0.f; }
  __syncthreads();

  { // row means of input fast
    const int r = t >> 4, c0 = t & 15;
    float s = 0.f;
    for (int c = c0; c < DD; c += 16) s += At[r * DD + c];
    atomicAdd(&rmean[r], s);
  }
  __syncthreads();

  const int wave = t >> 5, lane = t & 31;
  const int m = lane & 15, hi = lane >> 4;
  const int kh = hi * 2;                     // K pair select for f32 fragments
  v8f acc[4] = {};
  for (int kk = 0; kk < DD; kk += 4) {
    v2f a = *(const v2f*)&At[m * DD + kk + kh];
    #pragma unroll
    for (int j = 0; j < 4; ++j) {
      const int n = (wave * 4 + j) * 16 + m; // diffusion row = output column
      v2f bb = *(const v2f*)&diff[(size_t)n * DD + kk + kh];
      acc[j] = __builtin_amdgcn_wmma_f32_16x16x4_f32(
          false, a, false, bb, (short)0, acc[j], false, false);
    }
  }

  // epilogue: drift terms, noise, nan_to_num, row renorm, clip
  float ss[8];
  #pragma unroll
  for (int r = 0; r < 8; ++r) {
    const int mm = hi * 8 + r;
    const float mean = rmean[mm] * (1.f / DD);
    float a2 = 0.f;
    #pragma unroll
    for (int j = 0; j < 4; ++j) {
      const int nn = (wave * 4 + j) * 16 + m;
      const float f = At[mm * DD + nn];
      const size_t gi = (size_t)(row0 + mm) * DD + nn;
      float drift = acc[j][r] + CUBIC_C * tanhf(f - mean) + drive[gi] - LAMBDA_C * f;
      float s = f + DT_C * drift + NOISE_C * nz[gi];
      if (!__builtin_isfinite(s)) s = 0.f;
      acc[j][r] = s;
      a2 += s * s;
    }
    ss[r] = a2;
  }
  #pragma unroll
  for (int r = 0; r < 8; ++r) {
    #pragma unroll
    for (int msk = 8; msk >= 1; msk >>= 1) ss[r] += __shfl_xor(ss[r], msk, 32);
  }
  if (m == 0) {
    #pragma unroll
    for (int r = 0; r < 8; ++r) atomicAdd(&rnorm[hi * 8 + r], ss[r]);
  }
  __syncthreads();
  #pragma unroll
  for (int r = 0; r < 8; ++r) {
    const int mm = hi * 8 + r;
    const float inv = 1.f / (sqrtf(rnorm[mm]) + 1e-8f);
    #pragma unroll
    for (int j = 0; j < 4; ++j) {
      const int nn = (wave * 4 + j) * 16 + m;
      float v = acc[j][r] * inv;
      v = fminf(fmaxf(v, -10.f), 10.f);
      fout[(size_t)(row0 + mm) * DD + nn] = v;
    }
  }
}

// -------- kernel 3: combine + normalize + cast to bf16 --------
__global__ __launch_bounds__(256) void combine_kernel(
    const float* __restrict__ ff, const float* __restrict__ f0,
    const float* __restrict__ sl, __bf16* __restrict__ st)
{
  __shared__ float red[8];
  const int b = blockIdx.x, t = threadIdx.x;
  const size_t base = (size_t)b * DD;
  float c0 = 0.5f * (ff[base + t] + f0[base + t]) + 0.3f * sl[base + t];
  float c1 = 0.5f * (ff[base + t + 256] + f0[base + t + 256]) + 0.3f * sl[base + t + 256];
  float n = sqrtf(block_sum(c0 * c0 + c1 * c1, red));
  float inv = 1.f / (n + 1e-8f);
  st[base + t]       = (__bf16)(c0 * inv);
  st[base + t + 256] = (__bf16)(c1 * inv);
}

// -------- kernel 4: readout_w f32 -> bf16 with zero padding to VPAD rows --------
__global__ __launch_bounds__(256) void wconv_kernel(
    const float* __restrict__ w, __bf16* __restrict__ wb)
{
  const size_t i = ((size_t)blockIdx.x * 256 + threadIdx.x) * 4;
  const size_t lim = (size_t)VV * DD;        // multiple of 4
  float4 v = {0.f, 0.f, 0.f, 0.f};
  if (i < lim) v = *(const float4*)(w + i);
  wb[i + 0] = (__bf16)v.x;
  wb[i + 1] = (__bf16)v.y;
  wb[i + 2] = (__bf16)v.z;
  wb[i + 3] = (__bf16)v.w;
}

// -------- kernel 5: logits = (state @ W^T)/temp via v_wmma_f32_16x16x32_bf16 --------
// WG = 16 rows x 512 cols of V; 8 waves x 4 tiles; K = 512 in steps of 32.
__global__ __launch_bounds__(256) void readout_kernel(
    const __bf16* __restrict__ st, const __bf16* __restrict__ wb,
    const float* __restrict__ traw, float* __restrict__ out)
{
  const int t = threadIdx.x;
  const int wave = t >> 5, lane = t & 31;
  const int m = lane & 15, hi = lane >> 4;
  const int row0 = blockIdx.y * 16;
  const int tile0 = blockIdx.x * 32 + wave * 4;

  const float tr = *traw;                    // stable softplus
  const float sp = (tr > 0.f) ? (tr + log1pf(expf(-tr))) : log1pf(expf(tr));
  const float invT = 1.f / fmaxf(sp, 1e-6f);

  int tc[4];
  #pragma unroll
  for (int j = 0; j < 4; ++j) {              // clamp edge tiles (wave-uniform, EXEC stays full)
    int tj = tile0 + j;
    tc[j] = tj < NT16 ? tj : (NT16 - 1);
  }

  v8f acc[4] = {};
  const __bf16* arow = st + (size_t)(row0 + m) * DD;
  const int ka = hi * 8;                     // bf16 A: K in [8*hi,8*hi+8) and +16
  const int kb = hi * 8;                     // bf16 B: K in [8*hi,8*hi+8) and +16
  for (int kk = 0; kk < DD; kk += 32) {
    union { v16bf v; v8bf h[2]; } A;
    A.h[0] = *(const v8bf*)(arow + kk + ka);
    A.h[1] = *(const v8bf*)(arow + kk + 16 + ka);
    #pragma unroll
    for (int j = 0; j < 4; ++j) {
      const __bf16* brow = wb + (size_t)(tc[j] * 16 + m) * DD + kk + kb;
      union { v16bf v; v8bf h[2]; } Bm;
      Bm.h[0] = *(const v8bf*)(brow);
      Bm.h[1] = *(const v8bf*)(brow + 16);
      acc[j] = __builtin_amdgcn_wmma_f32_16x16x32_bf16(
          false, A.v, false, Bm.v, (short)0, acc[j], false, false);
    }
  }

  #pragma unroll
  for (int j = 0; j < 4; ++j) {
    const int tj = tile0 + j;
    if (tj >= NT16) continue;
    const int nn = tj * 16 + m;
    if (nn >= VV) continue;
    #pragma unroll
    for (int r = 0; r < 8; ++r) {
      const int mm = hi * 8 + r;
      float l = acc[j][r] * invT;
      if (l != l) l = 0.f;
      else if (__builtin_isinf(l)) l = (l > 0.f) ? 0.f : -10000.f;
      out[(size_t)(row0 + mm) * VV + nn] = l;
    }
  }
}

extern "C" void kernel_launch(void* const* d_in, const int* in_sizes, int n_in,
                              void* d_out, int out_size, void* d_ws, size_t ws_size,
                              hipStream_t stream) {
  (void)in_sizes; (void)n_in; (void)out_size; (void)ws_size;
  const int*   tok   = (const int*)d_in[0];
  const float* fast0 = (const float*)d_in[1];
  const float* slow0 = (const float*)d_in[2];
  const float* noise = (const float*)d_in[3];
  const float* embed = (const float*)d_in[4];
  const float* rw    = (const float*)d_in[5];
  const float* diff  = (const float*)d_in[6];
  const float* beta  = (const float*)d_in[7];
  const float* gamma = (const float*)d_in[8];
  const float* traw  = (const float*)d_in[9];
  float* out = (float*)d_out;

  char* ws = (char*)d_ws;                         // layout (MB offsets):
  float*  drive = (float*)(ws);                   // 0..4   : drive  [B,D] f32
  float*  fastA = (float*)(ws + (4ull  << 20));   // 4..8   : ping   [B,D] f32
  float*  fastB = (float*)(ws + (8ull  << 20));   // 8..12  : pong   [B,D] f32
  __bf16* stbf  = (__bf16*)(ws + (12ull << 20));  // 12..14 : state  [B,D] bf16
  __bf16* wbf   = (__bf16*)(ws + (14ull << 20));  // 14..~64: W      [VPAD,D] bf16

  drive_kernel<<<BD, 256, 0, stream>>>(tok, fast0, embed, beta, gamma, drive);
  wconv_kernel<<<(VPAD * DD) / (4 * 256), 256, 0, stream>>>(rw, wbf);

  const float* cur = fast0;
  for (int s = 0; s < NSTEP; ++s) {
    float* nxt = (s & 1) ? fastB : fastA;
    step_kernel<<<BD / 16, 256, 0, stream>>>(cur, nxt, diff, drive,
                                             noise + (size_t)s * BD * DD);
    cur = nxt;
  }

  combine_kernel<<<BD, 256, 0, stream>>>(cur, fast0, slow0, stbf);

  dim3 grid((NT16 + 31) / 32, BD / 16);           // 99 x 128
  readout_kernel<<<grid, 256, 0, stream>>>(stbf, wbf, traw, out);
}
